// HardAttention_24584392802432
// MI455X (gfx1250) — matrix operand order repaired and checked by reading, not compiled, not run on hardware
//
#include <hip/hip_runtime.h>

typedef __attribute__((ext_vector_type(16))) __bf16 v16bf;
typedef __attribute__((ext_vector_type(8)))  float  v8f;
typedef __attribute__((ext_vector_type(4)))  float  f32x4;
typedef __attribute__((ext_vector_type(8)))  float  f32x8;
typedef __attribute__((ext_vector_type(16))) float  f32x16;

// Problem constants (fixed by the reference): xs (B,Tx,d), ys (B,C,Ty,d)
constexpr int B_  = 16;
constexpr int TX  = 128;
constexpr int C_  = 64;
constexpr int TY  = 128;
constexpr int D_  = 768;
constexpr int KSTEP  = 32;           // K per v_wmma_f32_16x16x32_bf16
constexpr int NKSTEP = D_ / KSTEP;   // 24

// Pack 16 fp32 (four f32x4 chunks) into one v16bf operand. The shufflevector concat is
// register renaming only; the single vector fptrunc forces paired v_cvt_pk_bf16_f32
// with no v_mov_b16 packing fixups.
__device__ __forceinline__ v16bf cvt16(f32x4 r0, f32x4 r1, f32x4 r2, f32x4 r3)
{
    f32x8  lo  = __builtin_shufflevector(r0, r1, 0, 1, 2, 3, 4, 5, 6, 7);
    f32x8  hi  = __builtin_shufflevector(r2, r3, 0, 1, 2, 3, 4, 5, 6, 7);
    f32x16 all = __builtin_shufflevector(lo, hi, 0, 1, 2, 3, 4, 5, 6, 7,
                                                 8, 9, 10, 11, 12, 13, 14, 15);
    return __builtin_convertvector(all, v16bf);
}

__global__ __launch_bounds__(256)
void hardattn_bf16_wmma(const float* __restrict__ xs,
                        const float* __restrict__ ys,
                        float* __restrict__ out)
{
    const int b    = blockIdx.y;
    const int lane = threadIdx.x & 31;
    const int wave = threadIdx.x >> 5;
    const int ch   = wave >> 2;                 // which of the 2 c's this wave works on
    const int ws   = wave & 3;                  // row-strip pair: rows [ws*32, ws*32+32)
    const int c    = blockIdx.x * 2 + ch;
    const int grp  = lane >> 4;                 // half-wave 0/1
    const int l16  = lane & 15;

    // A rows: t = ws*32 + l16 (tile 0) and +16 (tile 1)
    const float* Xrow0 = xs + ((size_t)b * TX + (size_t)(ws * 32 + l16)) * D_;
    const float* Xrow1 = Xrow0 + (size_t)16 * D_;
    // B rows: s = st*16 + l16 of ys[b, c]
    const float* Yrow0 = ys + (((size_t)b * C_ + (size_t)c) * TY + (size_t)l16) * D_;

    v8f acc0[8] = {};    // rows [ws*32 .. +16), all 128 s columns
    v8f acc1[8] = {};    // rows [ws*32+16 .. +32)

    // A operands double-buffered across K iterations (parity constant-folds under
    // unroll-2) so the converts never WAR-collide with the previous iteration's WMMAs.
    v16bf abuf0[2], abuf1[2];

    #pragma unroll 2
    for (int ki = 0; ki < NKSTEP; ++ki) {
        const int kk  = ki * KSTEP;
        const int par = ki & 1;

        // ---- Two A tiles (16x32 bf16 each). Per-lane K chunks {c0..c0+7, c0+16..c0+23},
        //      c0 = kk + 8*grp (ISA 16-bit A layout).
        const int c0 = kk + 8 * grp;
        f32x4 x00 = *(const f32x4*)(Xrow0 + c0);
        f32x4 x01 = *(const f32x4*)(Xrow0 + c0 + 4);
        f32x4 x02 = *(const f32x4*)(Xrow0 + c0 + 16);
        f32x4 x03 = *(const f32x4*)(Xrow0 + c0 + 20);
        f32x4 x10 = *(const f32x4*)(Xrow1 + c0);
        f32x4 x11 = *(const f32x4*)(Xrow1 + c0 + 4);
        f32x4 x12 = *(const f32x4*)(Xrow1 + c0 + 16);
        f32x4 x13 = *(const f32x4*)(Xrow1 + c0 + 20);
        __builtin_prefetch(Xrow0 + c0 + KSTEP, 0, 1);
        __builtin_prefetch(Xrow1 + c0 + KSTEP, 0, 1);
        abuf0[par] = cvt16(x00, x01, x02, x03);
        abuf1[par] = cvt16(x10, x11, x12, x13);

        // ---- B tiles: lane holds column s = st*16 + l16, K chunks {kb..kb+7, kb+16..kb+23},
        //      kb = kk + 8*grp (same K order as A). Each B tile feeds TWO WMMAs.
        //      Triple-buffered bf16 staging: the cvt writing buffer (st%3) is 6 WMMAs
        //      behind the last WMMA that read it -> outside the 4-coexec hazard window.
        const int kb = kk + 8 * grp;
        f32x4 cur0 = *(const f32x4*)(Yrow0 + kb);
        f32x4 cur1 = *(const f32x4*)(Yrow0 + kb + 4);
        f32x4 cur2 = *(const f32x4*)(Yrow0 + kb + 16);
        f32x4 cur3 = *(const f32x4*)(Yrow0 + kb + 20);
        __builtin_prefetch(Yrow0 + kb + KSTEP, 0, 1);

        v16bf bv[3];

        #pragma unroll
        for (int st = 0; st < 8; ++st) {
            f32x4 n0, n1, n2, n3;
            if (st < 7) {   // issue next tile's loads before this tile's WMMAs
                const float* Yn = Yrow0 + (size_t)((st + 1) * 16) * D_ + kb;
                n0 = *(const f32x4*)(Yn);
                n1 = *(const f32x4*)(Yn + 4);
                n2 = *(const f32x4*)(Yn + 16);
                n3 = *(const f32x4*)(Yn + 20);
                __builtin_prefetch(Yn + KSTEP, 0, 1);
            }
            bv[st % 3] = cvt16(cur0, cur1, cur2, cur3);
            acc0[st] = __builtin_amdgcn_wmma_f32_16x16x32_bf16(
                           false, abuf0[par], false, bv[st % 3], (short)0, acc0[st],
                           false, false);
            acc1[st] = __builtin_amdgcn_wmma_f32_16x16x32_bf16(
                           false, abuf1[par], false, bv[st % 3], (short)0, acc1[st],
                           false, false);
            if (st < 7) { cur0 = n0; cur1 = n1; cur2 = n2; cur3 = n3; }
        }
    }

    // ---- Row-wise max over s (128 cols), then sum over the wave's 32 rows.
    // C/D layout: component r at lane L is (row = r + 8*(L>>4), col = L&15).
    v8f m0 = acc0[0], m1 = acc1[0];
    #pragma unroll
    for (int st = 1; st < 8; ++st) {
        #pragma unroll
        for (int r = 0; r < 8; ++r) {
            m0[r] = fmaxf(m0[r], acc0[st][r]);
            m1[r] = fmaxf(m1[r], acc1[st][r]);
        }
    }

    float rowsum = 0.0f;
    #pragma unroll
    for (int r = 0; r < 8; ++r) {
        float v0 = m0[r];
        float v1 = m1[r];
        v0 = fmaxf(v0, __shfl_xor(v0, 1, 16));
        v1 = fmaxf(v1, __shfl_xor(v1, 1, 16));
        v0 = fmaxf(v0, __shfl_xor(v0, 2, 16));
        v1 = fmaxf(v1, __shfl_xor(v1, 2, 16));
        v0 = fmaxf(v0, __shfl_xor(v0, 4, 16));
        v1 = fmaxf(v1, __shfl_xor(v1, 4, 16));
        v0 = fmaxf(v0, __shfl_xor(v0, 8, 16));
        v1 = fmaxf(v1, __shfl_xor(v1, 8, 16));
        rowsum += v0 + v1;      // row-maxes for this half-wave's 16 rows
    }
    rowsum += __shfl_xor(rowsum, 16, 32);   // combine half-waves -> all 32 rows

    __shared__ float part[8];               // [ch*4 + ws]
    if (lane == 0) part[wave] = rowsum;
    __syncthreads();
    if (threadIdx.x < 2) {                  // thread 0 -> c0, thread 1 -> c1
        float t = 0.0f;
        #pragma unroll
        for (int i = 0; i < 4; ++i) t += part[threadIdx.x * 4 + i];
        out[(size_t)b * C_ + (size_t)(blockIdx.x * 2 + threadIdx.x)] = t;
    }
}

extern "C" void kernel_launch(void* const* d_in, const int* in_sizes, int n_in,
                              void* d_out, int out_size, void* d_ws, size_t ws_size,
                              hipStream_t stream)
{
    const float* xs = (const float*)d_in[0];   // (16, 128, 768) fp32
    const float* ys = (const float*)d_in[1];   // (16, 64, 128, 768) fp32
    float* out = (float*)d_out;                // (16, 64) fp32

    dim3 grid(C_ / 2, B_);   // one workgroup per (b, pair-of-c)
    dim3 block(256);         // 8 wave32s: 2 c's x 4 row-strip-pairs
    hardattn_bf16_wmma<<<grid, block, 0, stream>>>(xs, ys, out);
}
